// PatchAttention_78675210928479
// MI455X (gfx1250) — matrix operand order repaired and checked by reading, not compiled, not run on hardware
//
#include <hip/hip_runtime.h>
#include <hip/hip_bf16.h>

// ---------------------------------------------------------------------------
// Types / WMMA helpers (CDNA5 gfx1250, wave32)
// ---------------------------------------------------------------------------
typedef __attribute__((ext_vector_type(16))) _Float16 v16h;
typedef __attribute__((ext_vector_type(8)))  float    v8f;
typedef __attribute__((ext_vector_type(4)))  int      v4i;

__device__ __forceinline__ v8f wmma_f16(v16h a, v16h b, v8f c) {
    return __builtin_amdgcn_wmma_f32_16x16x32_f16(
        false, a, false, b, (short)0, c, false, false);
}

// ---------------------------------------------------------------------------
// Async global->LDS copy (CDNA5). Guarded so missing builtins can't break
// the build; fallback is a plain 16B copy through VGPRs.
// Builtin signature (probed via hipcc diagnostic): (v4i AS1*, v4i AS3*, imm, imm)
// ---------------------------------------------------------------------------
#if defined(__has_builtin)
#if __has_builtin(__builtin_amdgcn_global_load_async_to_lds_b128)
#define HAVE_ASYNC_LDS 1
#endif
#endif
#ifndef HAVE_ASYNC_LDS
#define HAVE_ASYNC_LDS 0
#endif

__device__ __forceinline__ void async_b128(void* lds_dst, const void* gsrc) {
#if HAVE_ASYNC_LDS
    __builtin_amdgcn_global_load_async_to_lds_b128(
        (__attribute__((address_space(1))) v4i*)(v4i*)(void*)gsrc,
        (__attribute__((address_space(3))) v4i*)(v4i*)lds_dst, 0, 0);
#else
    *(int4*)lds_dst = *(const int4*)gsrc;
#endif
}

__device__ __forceinline__ void async_wait_all() {
#if HAVE_ASYNC_LDS
#if __has_builtin(__builtin_amdgcn_s_wait_asynccnt)
    __builtin_amdgcn_s_wait_asynccnt(0);
#else
    asm volatile("s_wait_asynccnt 0x0" ::: "memory");
#endif
#endif
}

// A fragment (16x32 f16) from row-major GLOBAL memory; tile origin (row0,k0).
__device__ __forceinline__ v16h load_a_g(const _Float16* __restrict__ A,
                                         int lda, int lane) {
    union { v16h h; unsigned int u[8]; } f;
    const _Float16* row = A + (size_t)(lane & 15) * lda;
    const int kb = (lane >= 16) ? 8 : 0;
#pragma unroll
    for (int j = 0; j < 8; ++j) {
        int k = ((j >= 4) ? 16 : 0) + kb + 2 * (j & 3);
        f.u[j] = *(const unsigned int*)(row + k);
    }
    return f.h;
}

// B fragment (32x16 f16) from "n-major" memory: element (k,n) at B[n*ldb+k].
__device__ __forceinline__ v16h load_b_g(const _Float16* __restrict__ B,
                                         int ldb, int lane) {
    union { v16h h; unsigned int u[8]; } f;
    const _Float16* col = B + (size_t)(lane & 15) * ldb + ((lane >= 16) ? 8 : 0);
#pragma unroll
    for (int j = 0; j < 8; ++j)
        f.u[j] = *(const unsigned int*)(col + ((j >= 4) ? 16 : 0) + 2 * (j & 3));
    return f.h;
}

// ---------------------------------------------------------------------------
// Problem constants
// ---------------------------------------------------------------------------
#define BB   8
#define CC   256
#define HH   160
#define WW   160
#define NPAT 400
#define EMB  256
#define KP   416      // padded attention K (400 -> 13 * 32)

// ---------------------------------------------------------------------------
// Kernel 1: weight/BN repack to f16
// conv_w [co][ci][3][3] f32 -> cwh [tap][co][ci] f16 (ci fastest: async-copy
// friendly, matches the conv LDS tile layout with no staging transpose)
// ---------------------------------------------------------------------------
__global__ void pack_kernel(const float* __restrict__ wq, const float* __restrict__ wk,
                            const float* __restrict__ convw,
                            const float* __restrict__ gamma, const float* __restrict__ beta,
                            const float* __restrict__ mean,  const float* __restrict__ var,
                            _Float16* __restrict__ wqh, _Float16* __restrict__ wkh,
                            _Float16* __restrict__ cwh, float* __restrict__ bnsc,
                            float* __restrict__ bnsh) {
    int idx = blockIdx.x * 256 + threadIdx.x;
    if (idx < 9 * 256 * 256) {
        int ci  = idx & 255;
        int co  = (idx >> 8) & 255;
        int tap = idx >> 16;                  // 0..8
        int dy = tap / 3, dx = tap - dy * 3;
        cwh[idx] = (_Float16)convw[(((co << 8) + ci) * 3 + dy) * 3 + dx];
    }
    if (idx < EMB * CC) {
        wqh[idx] = (_Float16)wq[idx];
        wkh[idx] = (_Float16)wk[idx];
    }
    if (idx < CC) {
        float sc = gamma[idx] * rsqrtf(var[idx] + 1e-3f);
        bnsc[idx] = sc;
        bnsh[idx] = beta[idx] - mean[idx] * sc;
    }
}

// ---------------------------------------------------------------------------
// Kernel 2: per-patch pooling + ReLU -> seq f16 [B*400][256]
// ---------------------------------------------------------------------------
__global__ void pool_kernel(const float* __restrict__ x, const float* __restrict__ pool_w,
                            _Float16* __restrict__ seq) {
    int bn = blockIdx.x;
    int b = bn / NPAT, n = bn - b * NPAT;
    int pyi = n / 20, pxi = n - pyi * 20;
    __shared__ float w[64];
    if (threadIdx.x < 64) w[threadIdx.x] = pool_w[n * 64 + threadIdx.x];
    __syncthreads();
    int c = threadIdx.x;
    const float* xp = x + ((((size_t)b * CC + c) * HH + pyi * 8) * WW + pxi * 8);
    float s = 0.f;
#pragma unroll
    for (int i = 0; i < 8; ++i)
#pragma unroll
        for (int j = 0; j < 8; ++j)
            s += xp[i * WW + j] * w[i * 8 + j];
    s = fmaxf(s, 0.f);
    seq[(size_t)bn * CC + c] = (_Float16)s;
}

// ---------------------------------------------------------------------------
// Kernel 3: Q/K projection GEMM.  M=3200, N=256, K=256, + bias.
// Block-uniform A tile (16x256) async-staged in LDS; 8 waves share it and
// cover 8 consecutive N tiles.
// ---------------------------------------------------------------------------
__global__ void qk_gemm(const _Float16* __restrict__ seq,
                        const _Float16* __restrict__ wqh, const _Float16* __restrict__ wkh,
                        const float* __restrict__ bq, const float* __restrict__ bk,
                        _Float16* __restrict__ qf, _Float16* __restrict__ kf) {
    __shared__ __align__(16) _Float16 lsA[16 * 256];
    int wave = threadIdx.x >> 5;
    int lane = threadIdx.x & 31;
    int tbase = blockIdx.x * 8;                  // block-uniform
    int which = (tbase >= 3200) ? 1 : 0;
    int t0 = tbase - which * 3200;
    int mt = t0 >> 4;                            // uniform across the block
    int nt = (t0 & 15) + wave;
    const _Float16* W = which ? wkh : wqh;
    const float* bias = which ? bk : bq;
    _Float16* out = which ? kf : qf;
    int m0 = mt * 16, n0 = nt * 16;

    // async-stage A tile [16][256] f16 = 8 KB (512 x 16B segments)
    for (int idx = threadIdx.x; idx < 512; idx += 256) {
        int row = idx >> 5, seg = idx & 31;
        async_b128(&lsA[row * 256 + seg * 8],
                   seq + (size_t)(m0 + row) * CC + seg * 8);
    }
    async_wait_all();
    __syncthreads();

    v8f acc = {};
#pragma unroll
    for (int k0 = 0; k0 < CC; k0 += 32) {
        union { v16h h; unsigned int u[8]; } fa;
        {
            int base = (lane & 15) * 256 + k0;
            int kb = (lane >= 16) ? 8 : 0;
#pragma unroll
            for (int j = 0; j < 8; ++j) {
                int k = ((j >= 4) ? 16 : 0) + kb + 2 * (j & 3);
                fa.u[j] = *(const unsigned int*)&lsA[base + k];
            }
        }
        v16h b = load_b_g(W + (size_t)n0 * CC + k0, CC, lane);
        acc = wmma_f16(fa.h, b, acc);
    }
    int n = n0 + (lane & 15);
    float bv = bias[n];
    int moff = (lane >= 16) ? 8 : 0;
#pragma unroll
    for (int r = 0; r < 8; ++r)
        out[(size_t)(m0 + r + moff) * CC + n] = (_Float16)(acc[r] + bv);
}

// ---------------------------------------------------------------------------
// Kernel 4: attention scores + softmax -> wts f16 [B][400][KP] (zero padded)
// Q tile async-staged in LDS (reused by all 8 waves x 25 column tiles).
// ---------------------------------------------------------------------------
__global__ void score_softmax(const _Float16* __restrict__ qf,
                              const _Float16* __restrict__ kf,
                              _Float16* __restrict__ wts) {
    int b = blockIdx.x / 25;
    int mt = blockIdx.x - b * 25;
    int wave = threadIdx.x >> 5, lane = threadIdx.x & 31;

    __shared__ float sc[16][400];
    __shared__ float red[16][17];
    __shared__ float rowv[16];
    __shared__ __align__(16) _Float16 lsQ[16 * 256];

    const _Float16* Q = qf + ((size_t)b * NPAT + mt * 16) * EMB;
    const _Float16* K = kf + (size_t)b * NPAT * EMB;

    for (int idx = threadIdx.x; idx < 512; idx += 256) {
        int row = idx >> 5, seg = idx & 31;
        async_b128(&lsQ[row * 256 + seg * 8], Q + (size_t)row * EMB + seg * 8);
    }
    async_wait_all();
    __syncthreads();

    for (int nt = wave; nt < 25; nt += 8) {
        v8f acc = {};
#pragma unroll
        for (int k0 = 0; k0 < EMB; k0 += 32) {
            union { v16h h; unsigned int u[8]; } fa;
            {
                int base = (lane & 15) * 256 + k0;
                int kb = (lane >= 16) ? 8 : 0;
#pragma unroll
                for (int j = 0; j < 8; ++j) {
                    int k = ((j >= 4) ? 16 : 0) + kb + 2 * (j & 3);
                    fa.u[j] = *(const unsigned int*)&lsQ[base + k];
                }
            }
            v16h bf = load_b_g(K + (size_t)(nt * 16) * EMB + k0, EMB, lane);
            acc = wmma_f16(fa.h, bf, acc);
        }
        int col = nt * 16 + (lane & 15);
        int moff = (lane >= 16) ? 8 : 0;
#pragma unroll
        for (int r = 0; r < 8; ++r)
            sc[r + moff][col] = acc[r] * 0.0625f;   // 1/sqrt(256)
    }
    __syncthreads();

    int row = threadIdx.x >> 4, j = threadIdx.x & 15;
    float pm = -3.4e38f;
    for (int c2 = j; c2 < 400; c2 += 16) pm = fmaxf(pm, sc[row][c2]);
    red[row][j] = pm;
    __syncthreads();
    if (j == 0) {
        float m = red[row][0];
#pragma unroll
        for (int i = 1; i < 16; ++i) m = fmaxf(m, red[row][i]);
        rowv[row] = m;
    }
    __syncthreads();
    float rm = rowv[row];
    float ps = 0.f;
    for (int c2 = j; c2 < 400; c2 += 16) {
        float e = __expf(sc[row][c2] - rm);
        sc[row][c2] = e;
        ps += e;
    }
    red[row][j] = ps;
    __syncthreads();
    if (j == 0) {
        float s = 0.f;
#pragma unroll
        for (int i = 0; i < 16; ++i) s += red[row][i];
        rowv[row] = s;
    }
    __syncthreads();
    float inv = 1.0f / rowv[row];
    _Float16* Wo = wts + ((size_t)b * NPAT + mt * 16) * KP;
    for (int c2 = j; c2 < 400; c2 += 16)
        Wo[row * KP + c2] = (_Float16)(sc[row][c2] * inv);
    Wo[row * KP + 400 + j] = (_Float16)0.f;
}

// ---------------------------------------------------------------------------
// Kernel 5: patch mix GEMM.  per batch: wts[400x416] x patches[416x16384].
// B panel [64 d x 416 k] built in LDS from x (implicit unfold + f32->f16, so
// not async-copyable); all 25 M tiles sweep against it.
// ---------------------------------------------------------------------------
__global__ void mix_gemm(const float* __restrict__ x,
                         const _Float16* __restrict__ wts,
                         _Float16* __restrict__ mixed) {
    int b = blockIdx.y;
    int d0 = blockIdx.x * 64;
    __shared__ __align__(16) _Float16 lb[64 * KP];   // [n_local][k]

    for (int idx = threadIdx.x; idx < 64 * KP; idx += 256) {
        int nl = idx / KP;
        int k = idx - nl * KP;
        float v = 0.f;
        if (k < 400) {
            int d = d0 + nl;
            int c = d >> 6, ph = (d >> 3) & 7, pw = d & 7;
            int pyi = k / 20, pxi = k - pyi * 20;
            v = x[(((size_t)b * CC + c) * HH + pyi * 8 + ph) * WW + pxi * 8 + pw];
        }
        lb[idx] = (_Float16)v;
    }
    __syncthreads();

    int wave = threadIdx.x >> 5, lane = threadIdx.x & 31;
    int mgrp = wave >> 2;
    int ntl  = wave & 3;

    for (int mt = mgrp; mt < 25; mt += 2) {
        v8f acc = {};
        const _Float16* A = wts + ((size_t)b * NPAT + mt * 16) * KP;
#pragma unroll
        for (int k0 = 0; k0 < KP; k0 += 32) {
            v16h a = load_a_g(A + k0, KP, lane);
            union { v16h h; unsigned int u[8]; } fb;
            int base = (ntl * 16 + (lane & 15)) * KP + k0 + ((lane >= 16) ? 8 : 0);
#pragma unroll
            for (int j = 0; j < 8; ++j)
                fb.u[j] = *(const unsigned int*)&lb[base + ((j >= 4) ? 16 : 0) + 2 * (j & 3)];
            acc = wmma_f16(a, fb.h, acc);
        }
        int d = d0 + ntl * 16 + (lane & 15);
        int c = d >> 6, ph = (d >> 3) & 7, pw = d & 7;
        int moff = (lane >= 16) ? 8 : 0;
#pragma unroll
        for (int r = 0; r < 8; ++r) {
            int m = mt * 16 + r + moff;
            int pyi = m / 20, pxi = m - pyi * 20;
            int yy = pyi * 8 + ph, xx = pxi * 8 + pw;
            mixed[(((size_t)b * HH + yy) * WW + xx) * CC + c] = (_Float16)acc[r];
        }
    }
}

// ---------------------------------------------------------------------------
// Kernel 6: 3x3 conv (implicit GEMM, K=2304) + BN + SiLU -> f32 out (NT).
// Per ci-chunk: async-stage input halo [6][18][32ci] + 9 weight taps [co][ci]
// directly into LDS (ASYNCcnt tracked), zero-filling only OOB halo entries.
// ---------------------------------------------------------------------------
__global__ void conv_bn_silu(const _Float16* __restrict__ mixed,
                             const _Float16* __restrict__ cwh,
                             const float* __restrict__ bnsc,
                             const float* __restrict__ bnsh,
                             float* __restrict__ out) {
    int b = blockIdx.y;
    int t = blockIdx.x;
    int cot = t & 3;
    int xt = (t >> 2) % 10;
    int yt = (t >> 2) / 10;
    int y0 = yt * 4, x0 = xt * 16, co0 = cot * 64;

    __shared__ __align__(16) _Float16 lin[6 * 18 * 32];  // [row][col][ci]
    __shared__ __align__(16) _Float16 lw[9][64 * 32];    // per tap: [co][ci]

    int wave = threadIdx.x >> 5, lane = threadIdx.x & 31;
    int mrow = wave >> 1;
    int nwv  = wave & 1;

    v8f acc0 = {}, acc1 = {};
    for (int c0 = 0; c0 < CC; c0 += 32) {
        __syncthreads();
        // input halo: 6*18 positions x 4 x 16B segments
        for (int idx = threadIdx.x; idx < 432; idx += 256) {
            int seg = idx & 3;
            int pos = idx >> 2;
            int col = pos % 18, row = pos / 18;
            int yy = y0 - 1 + row, xx = x0 - 1 + col;
            void* dst = &lin[(row * 18 + col) * 32 + seg * 8];
            if (yy >= 0 && yy < HH && xx >= 0 && xx < WW) {
                async_b128(dst, mixed + (((size_t)b * HH + yy) * WW + xx) * CC
                                      + c0 + seg * 8);
            } else {
                int4 z = {0, 0, 0, 0};
                *(int4*)dst = z;
            }
        }
        // weights: 9 taps x 64 co x 4 x 16B segments (contiguous ci in cwh)
        for (int idx = threadIdx.x; idx < 2304; idx += 256) {
            int seg = idx & 3;
            int co = (idx >> 2) & 63;
            int tap = idx >> 8;
            async_b128(&lw[tap][co * 32 + seg * 8],
                       cwh + ((size_t)tap * CC + co0 + co) * CC + c0 + seg * 8);
        }
        async_wait_all();
        __syncthreads();
#pragma unroll
        for (int tap = 0; tap < 9; ++tap) {
            int dy = tap / 3, dx = tap - dy * 3;
            union { v16h h; unsigned int u[8]; } fa;
            {
                int m = lane & 15;
                int kb = (lane >= 16) ? 8 : 0;
                int base = ((mrow + dy) * 18 + (m + dx)) * 32;
#pragma unroll
                for (int j = 0; j < 8; ++j) {
                    int k = ((j >= 4) ? 16 : 0) + kb + 2 * (j & 3);
                    fa.u[j] = *(const unsigned int*)&lin[base + k];
                }
            }
#pragma unroll
            for (int nt2 = 0; nt2 < 2; ++nt2) {
                union { v16h h; unsigned int u[8]; } fb;
                int n = nwv * 32 + nt2 * 16 + (lane & 15);
                int base = n * 32 + ((lane >= 16) ? 8 : 0);
#pragma unroll
                for (int j = 0; j < 8; ++j)
                    fb.u[j] = *(const unsigned int*)&lw[tap][base + ((j >= 4) ? 16 : 0) + 2 * (j & 3)];
                if (nt2 == 0) acc0 = wmma_f16(fa.h, fb.h, acc0);
                else          acc1 = wmma_f16(fa.h, fb.h, acc1);
            }
        }
    }
    int yy = y0 + mrow;
    int moff = (lane >= 16) ? 8 : 0;
#pragma unroll
    for (int nt2 = 0; nt2 < 2; ++nt2) {
        v8f acc = nt2 ? acc1 : acc0;
        int co = co0 + nwv * 32 + nt2 * 16 + (lane & 15);
        float s = bnsc[co], sh = bnsh[co];
#pragma unroll
        for (int r = 0; r < 8; ++r) {
            int xx = x0 + r + moff;
            float v = acc[r] * s + sh;
            v = v / (1.f + __expf(-v));
            __builtin_nontemporal_store(
                v, &out[(((size_t)b * CC + co) * HH + yy) * WW + xx]);
        }
    }
}

// ---------------------------------------------------------------------------
// Launch
// ---------------------------------------------------------------------------
extern "C" void kernel_launch(void* const* d_in, const int* in_sizes, int n_in,
                              void* d_out, int out_size, void* d_ws, size_t ws_size,
                              hipStream_t stream) {
    (void)in_sizes; (void)n_in; (void)out_size; (void)ws_size;
    const float* x      = (const float*)d_in[0];
    const float* pool_w = (const float*)d_in[1];
    const float* wq     = (const float*)d_in[2];
    const float* bq     = (const float*)d_in[3];
    const float* wk     = (const float*)d_in[4];
    const float* bk     = (const float*)d_in[5];
    const float* conv_w = (const float*)d_in[6];
    const float* gamma  = (const float*)d_in[7];
    const float* beta   = (const float*)d_in[8];
    const float* mean   = (const float*)d_in[9];
    const float* var    = (const float*)d_in[10];
    float* out = (float*)d_out;

    char* ws = (char*)d_ws;
    size_t off = 0;
    auto alloc = [&](size_t bytes) { char* p = ws + off; off += (bytes + 255) & ~(size_t)255; return p; };
    _Float16* seq   = (_Float16*)alloc((size_t)BB * NPAT * CC * 2);
    _Float16* qf    = (_Float16*)alloc((size_t)BB * NPAT * EMB * 2);
    _Float16* kf    = (_Float16*)alloc((size_t)BB * NPAT * EMB * 2);
    _Float16* wqh   = (_Float16*)alloc((size_t)EMB * CC * 2);
    _Float16* wkh   = (_Float16*)alloc((size_t)EMB * CC * 2);
    _Float16* cwh   = (_Float16*)alloc((size_t)9 * CC * CC * 2);
    _Float16* wts   = (_Float16*)alloc((size_t)BB * NPAT * KP * 2);
    float*    bnsc  = (float*)alloc(CC * 4);
    float*    bnsh  = (float*)alloc(CC * 4);
    _Float16* mixed = (_Float16*)alloc((size_t)BB * HH * WW * CC * 2);

    pack_kernel<<<(9 * 256 * 256 + 255) / 256, 256, 0, stream>>>(
        wq, wk, conv_w, gamma, beta, mean, var, wqh, wkh, cwh, bnsc, bnsh);

    pool_kernel<<<BB * NPAT, 256, 0, stream>>>(x, pool_w, seq);

    qk_gemm<<<800, 256, 0, stream>>>(seq, wqh, wkh, bq, bk, qf, kf);

    score_softmax<<<BB * 25, 256, 0, stream>>>(qf, kf, wts);

    {
        dim3 g(256, BB);
        mix_gemm<<<g, 256, 0, stream>>>(x, wts, mixed);
    }
    {
        dim3 g(1600, BB);
        conv_bn_silu<<<g, 256, 0, stream>>>(mixed, cwh, bnsc, bnsh, out);
    }
}
